// sCTRDT_Attention_39341900431769
// MI455X (gfx1250) — matrix operand order repaired and checked by reading, not compiled, not run on hardware
//
#include <hip/hip_runtime.h>

// sCTRDT attention for MI455X (gfx1250, wave32).
// All GEMM stages use v_wmma_f32_16x16x32_bf16. Shared operands (K/V tiles in
// attention, weight tiles in the GEMMs) are staged per-block into LDS with
// double-buffered GLOBAL_LOAD_ASYNC_TO_LDS_B128 (ASYNCcnt) when available.

#define Bsz 4
#define Ssz 1024
#define Hn  16
#define Dk  64
#define Dm  1024

typedef __attribute__((ext_vector_type(16))) __bf16 v16bf;
typedef __attribute__((ext_vector_type(8)))  float  v8f;
typedef int v4i_ __attribute__((vector_size(16)));

union Frag16 { v16bf v; unsigned int u[8]; };

#define AS1 __attribute__((address_space(1)))
#define AS3 __attribute__((address_space(3)))

#if defined(__has_builtin)
#if __has_builtin(__builtin_amdgcn_global_load_async_to_lds_b128)
#define HAVE_ASYNC 1
#endif
#endif
#ifndef HAVE_ASYNC
#define HAVE_ASYNC 0
#endif

// 16-byte global -> LDS copy (async when the toolchain exposes the builtin).
// Builtin signature (from toolchain diagnostic): (int4 AS1*, int4 AS3*, Ii, Ii).
__device__ __forceinline__ void async_cp16(unsigned short* lds,
                                           const unsigned short* g) {
#if HAVE_ASYNC
  __builtin_amdgcn_global_load_async_to_lds_b128(
      (AS1 v4i_*)g, (AS3 v4i_*)lds, 0, 0);
#else
  *(uint4*)lds = *(const uint4*)g;
#endif
}

__device__ __forceinline__ void wait_async() {
#if HAVE_ASYNC
  asm volatile("s_wait_asynccnt 0" ::: "memory");
#endif
}

__device__ __forceinline__ unsigned short f32_to_bf16(float f) {
  unsigned int u = __float_as_uint(f);
  u += 0x7FFFu + ((u >> 16) & 1u);            // round-to-nearest-even
  return (unsigned short)(u >> 16);
}

__device__ __forceinline__ v8f wmma_bf16(v16bf a, v16bf b, v8f c) {
  return __builtin_amdgcn_wmma_f32_16x16x32_bf16(false, a, false, b, (short)0, c,
                                                 false, false);
}

// A-operand (16x32 bf16) per-lane K pattern (contiguous, matches B fills):
// lanes 0-15: K = {0..15}; lanes 16-31: K = {16..31}. Pairs (2r,2r+1) -> one dword.
__device__ __forceinline__ int a_kpat(int r, int hi8) {
  return 2 * r + 2 * hi8;
}

// ---------------------------------------------------------------- t_scaled ---
__global__ void tscale_kernel(const float* __restrict__ t_raw,
                              float* __restrict__ tsc) {
  __shared__ float smn[256], smx[256];
  const int b = blockIdx.x, tid = threadIdx.x;
  const float* t = t_raw + b * Ssz;
  float mn = 1e30f, mx = -1e30f;
  for (int i = tid; i < Ssz; i += 256) {
    float v = t[i];
    mn = fminf(mn, v); mx = fmaxf(mx, v);
  }
  smn[tid] = mn; smx[tid] = mx;
  __syncthreads();
  for (int s = 128; s > 0; s >>= 1) {
    if (tid < s) {
      smn[tid] = fminf(smn[tid], smn[tid + s]);
      smx[tid] = fmaxf(smx[tid], smx[tid + s]);
    }
    __syncthreads();
  }
  const float tmin = smn[0];
  const float inv = 1000.0f / (smx[0] - tmin + 1e-6f);
  for (int i = tid; i < Ssz; i += 256) tsc[b * Ssz + i] = (t[i] - tmin) * inv;
}

// ------------------------------------------------------------- f32 -> bf16 ---
__global__ void cvt_bf16_kernel(const float* __restrict__ src,
                                unsigned short* __restrict__ dst, int n) {
  for (int i = blockIdx.x * blockDim.x + threadIdx.x; i < n;
       i += gridDim.x * blockDim.x)
    dst[i] = f32_to_bf16(src[i]);
}

// ------------------------------------------------------------ WMMA GEMM ------
// C[M,N] = A_bf[M,1024] @ W_bf[N,1024]^T (+bias, +epilogue).
// Block = 8 waves; the 8 waves share one 64-wide N tile (tiles_m % 8 == 0), so
// the weight k-slice (64x32 bf16 = 4KB) is async-staged once per block into a
// double-buffered LDS tile. One wave computes a 16x64 output tile.
// mode 0: write f32 (+bo) to outf.
// mode 1: fused QKV (N=3072): bias + RoPE(q,k) + scatter to
//   q_bf/k_bf [B,H,S,Dk] and vT_bf [B,H,Dk,S] (all bf16).
__global__ void gemm_bf16_kernel(const unsigned short* __restrict__ A,
                                 const unsigned short* __restrict__ W,
                                 int tiles_m, int tiles_n, int mode,
                                 const float* __restrict__ b0,
                                 const float* __restrict__ b1,
                                 const float* __restrict__ b2,
                                 const float* __restrict__ tsc,
                                 float* __restrict__ outf,
                                 unsigned short* __restrict__ q_bf,
                                 unsigned short* __restrict__ k_bf,
                                 unsigned short* __restrict__ vT_bf) {
  __shared__ __align__(16) unsigned short wbuf[2][64 * 32];  // 2 x 4KB

  const int tid = threadIdx.x;
  const int wave = tid >> 5, lane = tid & 31;
  const int tile = blockIdx.x * 8 + wave;
  const int tm = tile % tiles_m, tn = tile / tiles_m;  // tn uniform per block
  (void)tiles_n;
  const int nlane = lane & 15, hi = lane >> 4;
  const int koff8 = hi * 8, koff16 = hi * 16, roff = hi * 8;

  const unsigned short* ar = A + (size_t)(tm * 16 + nlane) * Dm;
  const unsigned short* wrow = W + (size_t)(tn * 64 + (tid >> 2)) * Dm + (tid & 3) * 8;
  unsigned short* wdst = &wbuf[0][0] + (tid >> 2) * 32 + (tid & 3) * 8;

  v8f c[4] = {{}, {}, {}, {}};

  // prologue: stage k0 = 0
  async_cp16(wdst, wrow);
  wait_async();
  __syncthreads();

  int cur = 0;
  for (int k0 = 0; k0 < Dm; k0 += 32) {
    const int nxt = cur ^ 1;
    if (k0 + 32 < Dm)                       // prefetch next weight k-slice
      async_cp16(wdst + nxt * (64 * 32), wrow + k0 + 32);

    Frag16 a;
#pragma unroll
    for (int r = 0; r < 8; ++r)
      a.u[r] = *(const unsigned int*)(ar + k0 + a_kpat(r, koff8));

    const unsigned short* wb = &wbuf[cur][0];
#pragma unroll
    for (int t = 0; t < 4; ++t) {
      Frag16 bw;
      bw.v = *(const v16bf*)(wb + (t * 16 + nlane) * 32 + koff16);
      c[t] = wmma_bf16(a.v, bw.v, c[t]);
    }
    wait_async();
    __syncthreads();
    cur = nxt;
  }

  if (mode == 0) {
#pragma unroll
    for (int t = 0; t < 4; ++t) {
      const int col = tn * 64 + t * 16 + nlane;
      const float bc = b0[col];
#pragma unroll
      for (int i = 0; i < 8; ++i) {
        const int Mrow = tm * 16 + roff + i;
        outf[(size_t)Mrow * Dm + col] = c[t][i] + bc;
      }
    }
  } else {
    const int o_base = tn * 64;
    const int proj = o_base >> 10;            // 0=q 1=k 2=v
    const int hh = (o_base & 1023) >> 6;      // head (tile spans one head)
    const float* bias = (proj == 0) ? b0 : (proj == 1 ? b1 : b2);
    float trow[8];
#pragma unroll
    for (int i = 0; i < 8; ++i) trow[i] = tsc[tm * 16 + roff + i];
#pragma unroll
    for (int t = 0; t < 4; ++t) {
      const int d = t * 16 + nlane;
      const float bc = bias[hh * 64 + d];
      if (proj < 2) {
        // RoPE: angle = t_scaled * base^(-(d&~1)/64); pair partner in lane^1.
        const float freq = __expf(-((float)(d & ~1) * (1.0f / 64.0f)) * 9.2103403720f);
        unsigned short* dst = (proj == 0) ? q_bf : k_bf;
#pragma unroll
        for (int i = 0; i < 8; ++i) {
          const float val = c[t][i] + bc;
          const float par = __shfl_xor(val, 1);
          float sn, cs;
          __sincosf(trow[i] * freq, &sn, &cs);
          const float rot = (d & 1) ? par : -par;
          const float outv = val * cs + rot * sn;
          const int Mrow = tm * 16 + roff + i;
          const int bb = Mrow >> 10, s = Mrow & 1023;
          dst[(((size_t)bb * Hn + hh) * Ssz + s) * Dk + d] = f32_to_bf16(outv);
        }
      } else {
#pragma unroll
        for (int i = 0; i < 8; ++i) {
          const float val = c[t][i] + bc;
          const int Mrow = tm * 16 + roff + i;
          const int bb = Mrow >> 10, s = Mrow & 1023;
          vT_bf[(((size_t)bb * Hn + hh) * Dk + d) * Ssz + s] = f32_to_bf16(val);
        }
      }
    }
  }
}

// ------------------------------------------------- flash attention kernel ----
// grid: (S/128, H, B); block 256 = 8 waves; wave owns 16 query rows, streams
// all 1024 keys in 32-key tiles with online softmax. K (32x64) and V^T (64x32)
// tiles are shared by all 8 waves -> async-staged per block into LDS with
// double buffering; the copy of tile j+1 overlaps WMMA work on tile j.
__global__ void attn_kernel(const unsigned short* __restrict__ q_bf,
                            const unsigned short* __restrict__ k_bf,
                            const unsigned short* __restrict__ vT_bf,
                            const float* __restrict__ tsc,
                            const float* __restrict__ err,
                            const float* __restrict__ lambda_h,
                            const float* __restrict__ Wg,
                            const float* __restrict__ bg,
                            unsigned short* __restrict__ attn_bf) {
  __shared__ float ts_s[Ssz];
  __shared__ float es_s[Ssz];
  __shared__ __align__(16) unsigned short p_lds[8 * 16 * 32];   // 8KB
  __shared__ __align__(16) unsigned short kbuf[2][32 * 64];     // 2 x 4KB
  __shared__ __align__(16) unsigned short vbuf[2][64 * 32];     // 2 x 4KB

  const int b = blockIdx.z, h = blockIdx.y;
  const int tid = threadIdx.x, wave = tid >> 5, lane = tid & 31;
  for (int i = tid; i < Ssz; i += 256) {
    ts_s[i] = tsc[b * Ssz + i];
    es_s[i] = err[b * Ssz + i];
  }

  const float lam = log1pf(__expf(lambda_h[h]));   // softplus
  const float wg0 = Wg[2 * h], wg1 = Wg[2 * h + 1], bgh = bg[h];
  const int q0 = blockIdx.x * 128 + wave * 16;
  const int nlane = lane & 15, hi = lane >> 4;
  const int koff8 = hi * 8, koff16 = hi * 16, roff = hi * 8;

  const size_t bh = (size_t)(b * Hn + h);
  const unsigned short* qp = q_bf + bh * Ssz * Dk;
  const unsigned short* kp = k_bf + bh * Ssz * Dk;
  const unsigned short* vp = vT_bf + bh * Dk * Ssz;

  // per-thread staging slots: one 16B chunk of K, one 16B chunk of V^T
  const unsigned short* kg = kp + (size_t)(tid >> 3) * Dk + (tid & 7) * 8;
  unsigned short* kd = &kbuf[0][0] + (tid >> 3) * 64 + (tid & 7) * 8;
  const unsigned short* vg = vp + (size_t)(tid >> 2) * Ssz + (tid & 3) * 8;
  unsigned short* vd = &vbuf[0][0] + (tid >> 2) * 32 + (tid & 3) * 8;

  // Q fragments (d 0..31 and 32..63), loaded once.
  Frag16 aq0, aq1;
  {
    const unsigned short* qr = qp + (size_t)(q0 + nlane) * Dk;
#pragma unroll
    for (int r = 0; r < 8; ++r) {
      const int kpat = a_kpat(r, koff8);
      aq0.u[r] = *(const unsigned int*)(qr + kpat);
      aq1.u[r] = *(const unsigned int*)(qr + 32 + kpat);
    }
  }

  // prologue: stage key tile j0 = 0 (also covers the ts_s/es_s fill barrier)
  async_cp16(kd, kg);
  async_cp16(vd, vg);
  wait_async();
  __syncthreads();

  float tq[8], gaq[8], m_run[8], l_run[8];
#pragma unroll
  for (int i = 0; i < 8; ++i) {
    const int r = q0 + roff + i;
    tq[i] = ts_s[r];
    gaq[i] = es_s[r] * wg0;
    m_run[i] = -1e30f;
    l_run[i] = 0.0f;
  }
  v8f oacc[4] = {{}, {}, {}, {}};
  unsigned short* pl = p_lds + wave * (16 * 32);

  int cur = 0;
  for (int j0 = 0; j0 < Ssz; j0 += 32) {
    const int nxt = cur ^ 1;
    if (j0 + 32 < Ssz) {                     // prefetch next K/V tile
      async_cp16(kd + nxt * (32 * 64), kg + (size_t)(j0 + 32) * Dk);
      async_cp16(vd + nxt * (64 * 32), vg + (j0 + 32));
    }
    const unsigned short* kt = &kbuf[cur][0];
    const unsigned short* vt = &vbuf[cur][0];

    // K fragments: B^T rows = key rows (contiguous 32B/lane from LDS).
    Frag16 k00, k01, k10, k11;
    const unsigned short* kr0 = kt + nlane * Dk + koff16;
    const unsigned short* kr1 = kr0 + 16 * Dk;
    k00.v = *(const v16bf*)(kr0);       k01.v = *(const v16bf*)(kr0 + 32);
    k10.v = *(const v16bf*)(kr1);       k11.v = *(const v16bf*)(kr1 + 32);

    v8f c0 = {}, c1 = {};
    c0 = wmma_bf16(aq0.v, k00.v, c0);
    c0 = wmma_bf16(aq1.v, k01.v, c0);
    c1 = wmma_bf16(aq0.v, k10.v, c1);
    c1 = wmma_bf16(aq1.v, k11.v, c1);

    const float tj0 = ts_s[j0 + nlane],       tj1 = ts_s[j0 + 16 + nlane];
    const float gb0 = es_s[j0 + nlane] * wg1, gb1 = es_s[j0 + 16 + nlane] * wg1;

#pragma unroll
    for (int i = 0; i < 8; ++i) {
      const float z0 = gaq[i] + gb0 + bgh, z1 = gaq[i] + gb1 + bgh;
      float s0 = c0[i] * 0.125f - lam * fabsf(tq[i] - tj0)
               + __logf(1.0f / (1.0f + __expf(-z0)) + 1e-9f);
      float s1 = c1[i] * 0.125f - lam * fabsf(tq[i] - tj1)
               + __logf(1.0f / (1.0f + __expf(-z1)) + 1e-9f);
      // row max across the 16 lanes of this half (row lives in one half)
      float mx = fmaxf(s0, s1);
      mx = fmaxf(mx, __shfl_xor(mx, 1));
      mx = fmaxf(mx, __shfl_xor(mx, 2));
      mx = fmaxf(mx, __shfl_xor(mx, 4));
      mx = fmaxf(mx, __shfl_xor(mx, 8));
      const float mnew = fmaxf(m_run[i], mx);
      const float sc = __expf(m_run[i] - mnew);
      const float p0 = __expf(s0 - mnew), p1 = __expf(s1 - mnew);
      float rs = p0 + p1;
      rs += __shfl_xor(rs, 1);
      rs += __shfl_xor(rs, 2);
      rs += __shfl_xor(rs, 4);
      rs += __shfl_xor(rs, 8);
      l_run[i] = l_run[i] * sc + rs;
      m_run[i] = mnew;
      oacc[0][i] *= sc; oacc[1][i] *= sc; oacc[2][i] *= sc; oacc[3][i] *= sc;
      const int M = roff + i;                 // row of P tile
      pl[M * 32 + nlane]      = f32_to_bf16(p0);
      pl[M * 32 + 16 + nlane] = f32_to_bf16(p1);
    }
    // wave-local LDS transpose handoff (DS ops are in-order per wave)
    asm volatile("s_wait_dscnt 0" ::: "memory");

    Frag16 ap;                                // P as A-operand (16x32)
#pragma unroll
    for (int r = 0; r < 8; ++r)
      ap.u[r] = *(const unsigned int*)&pl[nlane * 32 + a_kpat(r, koff8)];

#pragma unroll
    for (int t = 0; t < 4; ++t) {             // V^T rows = d, contiguous keys
      Frag16 vf;
      vf.v = *(const v16bf*)(vt + (t * 16 + nlane) * 32 + koff16);
      oacc[t] = wmma_bf16(ap.v, vf.v, oacc[t]);
    }
    wait_async();
    __syncthreads();
    cur = nxt;
  }

  // epilogue: normalize and scatter to [b, s, h*64+d] bf16
#pragma unroll
  for (int i = 0; i < 8; ++i) {
    const float inv = 1.0f / l_run[i];
    const int q = q0 + roff + i;
#pragma unroll
    for (int t = 0; t < 4; ++t) {
      const int d = t * 16 + nlane;
      attn_bf[((size_t)b * Ssz + q) * Dm + h * Dk + d] =
          f32_to_bf16(oacc[t][i] * inv);
    }
  }
}

// ----------------------------------------------------------------- driver ----
extern "C" void kernel_launch(void* const* d_in, const int* in_sizes, int n_in,
                              void* d_out, int out_size, void* d_ws, size_t ws_size,
                              hipStream_t stream) {
  (void)in_sizes; (void)n_in; (void)out_size; (void)ws_size;
  const float* x        = (const float*)d_in[0];
  const float* t_raw    = (const float*)d_in[1];
  const float* error    = (const float*)d_in[2];
  const float* Wq       = (const float*)d_in[3];
  const float* bq       = (const float*)d_in[4];
  const float* Wk       = (const float*)d_in[5];
  const float* bk       = (const float*)d_in[6];
  const float* Wv       = (const float*)d_in[7];
  const float* bv       = (const float*)d_in[8];
  const float* Wo       = (const float*)d_in[9];
  const float* bo       = (const float*)d_in[10];
  const float* lambda_h = (const float*)d_in[11];
  const float* Wg       = (const float*)d_in[12];
  const float* bg       = (const float*)d_in[13];
  float* outf = (float*)d_out;

  char* ws = (char*)d_ws;
  auto alloc = [&](size_t bytes) -> char* {
    char* p = ws;
    ws += (bytes + 255) & ~(size_t)255;
    return p;
  };
  unsigned short* x_bf    = (unsigned short*)alloc((size_t)Bsz * Ssz * Dm * 2);
  unsigned short* wqkv_bf = (unsigned short*)alloc((size_t)3 * Dm * Dm * 2);
  unsigned short* wo_bf   = (unsigned short*)alloc((size_t)Dm * Dm * 2);
  float*          tsc     = (float*)alloc((size_t)Bsz * Ssz * 4);
  unsigned short* q_bf    = (unsigned short*)alloc((size_t)Bsz * Hn * Ssz * Dk * 2);
  unsigned short* k_bf    = (unsigned short*)alloc((size_t)Bsz * Hn * Ssz * Dk * 2);
  unsigned short* vT_bf   = (unsigned short*)alloc((size_t)Bsz * Hn * Dk * Ssz * 2);
  unsigned short* attn_bf = (unsigned short*)alloc((size_t)Bsz * Ssz * Dm * 2);

  const int NW = Dm * Dm;                       // 1M elems per weight
  cvt_bf16_kernel<<<2048, 256, 0, stream>>>(x,  x_bf, Bsz * Ssz * Dm);
  cvt_bf16_kernel<<<1024, 256, 0, stream>>>(Wq, wqkv_bf,          NW);
  cvt_bf16_kernel<<<1024, 256, 0, stream>>>(Wk, wqkv_bf + NW,     NW);
  cvt_bf16_kernel<<<1024, 256, 0, stream>>>(Wv, wqkv_bf + 2 * NW, NW);
  cvt_bf16_kernel<<<1024, 256, 0, stream>>>(Wo, wo_bf,            NW);
  tscale_kernel<<<Bsz, 256, 0, stream>>>(t_raw, tsc);

  // fused QKV projection + RoPE + scatter: M=4096 (256 tiles), N=3072 (48 tiles)
  gemm_bf16_kernel<<<(256 * 48) / 8, 256, 0, stream>>>(
      x_bf, wqkv_bf, 256, 48, 1, bq, bk, bv, tsc,
      nullptr, q_bf, k_bf, vT_bf);

  attn_kernel<<<dim3(Ssz / 128, Hn, Bsz), 256, 0, stream>>>(
      q_bf, k_bf, vT_bf, tsc, error, lambda_h, Wg, bg, attn_bf);

  // output projection: M=4096 (256 tiles), N=1024 (16 tiles), f32 out
  gemm_bf16_kernel<<<(256 * 16) / 8, 256, 0, stream>>>(
      attn_bf, wo_bf, 256, 16, 0, bo, nullptr, nullptr, nullptr,
      outf, nullptr, nullptr, nullptr);
}
